// EncodeProcessDecode_74852690035243
// MI455X (gfx1250) — matrix operand order repaired and checked by reading, not compiled, not run on hardware
//
#include <hip/hip_runtime.h>

// ---------------------------------------------------------------------------
// EncodeProcessDecode GNN for MI455X (gfx1250, wave32, WMMA).
// GEMMs: v_wmma_f32_16x16x32_bf16 (fp32 accumulate). LN / ReLU / residual /
// activations stay fp32. Concat inputs (3L edge-MLP, 2L node-MLP) are
// gathered on the fly per 32-wide K chunk - never materialized.
//
// Weights are pre-converted once per launch to bf16 in the exact B-fragment
// LDS image layout (swizzle_w_kernel), so each layer kernel stages its whole
// weight image into LDS with async global->LDS b128 copies (ASYNCcnt) and
// then runs a barrier-free, fully unrolled WMMA loop.
//
// Workspace (floats): n[N*128] | agg[N*128] | e[E*128] | e_new[E*128] |
// h[E*128] | bf16 weight images (~3.3 MB). ~823 MB total assumed in d_ws.
// ---------------------------------------------------------------------------

typedef __attribute__((ext_vector_type(16))) __bf16 v16bf;
typedef __attribute__((ext_vector_type(8)))  float  v8f;
typedef __attribute__((ext_vector_type(4)))  float  f32x4;

// ---------------------------------------------------------------------------
// CDNA5 async global->LDS copy (16B per lane), tracked by ASYNCcnt.
// ---------------------------------------------------------------------------
__device__ __forceinline__ void async_copy_b128(unsigned lds_byte_addr, const void* gsrc) {
  asm volatile("global_load_async_to_lds_b128 %0, %1, off"
               :: "v"(lds_byte_addr), "v"(gsrc) : "memory");
}
__device__ __forceinline__ void wait_async_all() {
  asm volatile("s_wait_asynccnt 0" ::: "memory");
}

// ---------------------------------------------------------------------------
// One-time weight prep: fp32 [KDIM,128] -> bf16 image in B-fragment order:
//   out[((c*8 + nt)*32 + l)*16 + j] = bf16( W[(c*32 + (l>>4)*8 + (j&7) + (j>>3)*16)*128
//                                            + nt*16 + (l&15)] )
// Each thread writes 8 consecutive bf16 (16B contiguous stores).
// ---------------------------------------------------------------------------
__global__ __launch_bounds__(256)
void swizzle_w_kernel(const float* __restrict__ W, __bf16* __restrict__ out, int kdim)
{
  int g = blockIdx.x * 256 + threadIdx.x;        // 8 elements per thread
  int total = kdim * 16;                          // (kdim*128)/8
  if (g >= total) return;
  int base = g * 8;
  #pragma unroll
  for (int q = 0; q < 8; ++q) {
    int i    = base + q;
    int j    = i & 15;
    int l    = (i >> 4) & 31;
    int nt   = (i >> 9) & 7;
    int c    = i >> 12;
    int krow = ((l >> 4) << 3) + (j & 7) + ((j >> 3) << 4);
    int col  = nt * 16 + (l & 15);
    out[i] = (__bf16)W[(size_t)(c * 32 + krow) * 128 + col];
  }
}

// ---------------------------------------------------------------------------
// Y[M,128] = act( LN( X[M,KDIM] @ W[KDIM,128] + bias ) )
//   X assembled from up to three 128-wide segments:
//     cols [0,128):X0[idx0[row]]  [128,256):X1[idx1[row]]  [256,384):X2[row]
//   Y2 != null  : Y = raw output, Y2 = resid + output (fused residual stream)
//   resid != null (Y2 null): Y = resid + output
// Block = 256 threads = 8 waves; wave w owns rows [tile+16w, tile+16w+16),
// computing all 8 N-tiles of the 16x128 strip with bf16 WMMA.
// Whole bf16 weight image lives in LDS (staged once, async): no loop barriers.
// ---------------------------------------------------------------------------
template<int KDIM, bool LN, bool RELU>
__global__ __launch_bounds__(256)
void mlp_layer_kernel(const float* __restrict__ X0, const int* __restrict__ idx0,
                      const float* __restrict__ X1, const int* __restrict__ idx1,
                      const float* __restrict__ X2,
                      const __bf16* __restrict__ Wsw,  // swizzled bf16 image [KDIM*128]
                      const float* __restrict__ bias,
                      const float* __restrict__ gamma, const float* __restrict__ beta,
                      float* __restrict__ Y, const float* __restrict__ resid,
                      float* __restrict__ Y2, int M)
{
  constexpr int NCHUNK = KDIM / 32;
  __shared__ __align__(32) __bf16 Blds[KDIM * 128]; // 32/64/96 KB

  const int tid  = threadIdx.x;
  const int wave = tid >> 5;
  const int lane = tid & 31;
  const int l16  = lane & 15;
  const int lh   = lane >> 4;                 // lane half (K-group select)
  const int tileRow = blockIdx.x * 128 + wave * 16;

  // ---- stage the whole weight image into LDS via async b128 copies ----
  {
    constexpr int PKTS = KDIM / 16;           // 16B packets per thread
    unsigned ldsbase = (unsigned)(size_t)&Blds[0];
    const char* gsrc = (const char*)Wsw;
    #pragma unroll
    for (int p = 0; p < PKTS; ++p) {
      unsigned off = (unsigned)(p * 4096 + tid * 16);
      async_copy_b128(ldsbase + off, gsrc + off);
    }
    wait_async_all();
  }

  int arow = tileRow + l16;                   // A-matrix row for this lane
  if (arow >= M) arow = M - 1;                // clamp tail (stores guarded)
  const int r0 = idx0 ? idx0[arow] : arow;
  const int r1 = (KDIM > 128) ? (idx1 ? idx1[arow] : arow) : 0;

  v8f acc[8];
  #pragma unroll
  for (int nt = 0; nt < 8; ++nt)
    #pragma unroll
    for (int r = 0; r < 8; ++r) acc[nt][r] = 0.0f;

  __syncthreads();                            // LDS image visible to all waves

  #pragma unroll
  for (int c = 0; c < NCHUNK; ++c) {
    // ---- A fragment (16x32 bf16) straight from global, per ISA layout ----
    const int colBase = c * 32;
    const float* src; int srow;
    if (KDIM == 128) { src = X0; srow = r0; }
    else {
      const int seg = colBase >> 7;           // 32 | 128: chunk stays in one segment
      if (seg == 0)      { src = X0; srow = r0; }
      else if (seg == 1) { src = X1; srow = r1; }
      else               { src = X2; srow = arow; }
    }
    const float* p = src + (size_t)srow * 128 + (colBase & 127) + lh * 8;
    f32x4 a0 = *(const f32x4*)(p);            // K = lh*8 + 0..3
    f32x4 a1 = *(const f32x4*)(p + 4);        // K = lh*8 + 4..7
    f32x4 a2 = *(const f32x4*)(p + 16);       // K = 16 + lh*8 + 0..3
    f32x4 a3 = *(const f32x4*)(p + 20);       // K = 16 + lh*8 + 4..7
    v16bf a;
    a[0]  = (__bf16)a0.x; a[1]  = (__bf16)a0.y; a[2]  = (__bf16)a0.z; a[3]  = (__bf16)a0.w;
    a[4]  = (__bf16)a1.x; a[5]  = (__bf16)a1.y; a[6]  = (__bf16)a1.z; a[7]  = (__bf16)a1.w;
    a[8]  = (__bf16)a2.x; a[9]  = (__bf16)a2.y; a[10] = (__bf16)a2.z; a[11] = (__bf16)a2.w;
    a[12] = (__bf16)a3.x; a[13] = (__bf16)a3.y; a[14] = (__bf16)a3.z; a[15] = (__bf16)a3.w;

    // ---- 8 WMMAs: one per 16-wide N tile ----
    #pragma unroll
    for (int nt = 0; nt < 8; ++nt) {
      v16bf b = *(const v16bf*)&Blds[((c * 8 + nt) * 32 + lane) * 16];
      acc[nt] = __builtin_amdgcn_wmma_f32_16x16x32_bf16(
          false, a, false, b, (short)0, acc[nt], false, false);
    }
  }

  // ---- epilogue: bias, LayerNorm (in-register shuffle reduce), ReLU ----
  float bv[8], gv[8], bev[8];
  #pragma unroll
  for (int nt = 0; nt < 8; ++nt) {
    int n = nt * 16 + l16;
    bv[nt] = bias[n];
    if (LN) { gv[nt] = gamma[n]; bev[nt] = beta[n]; }
  }
  #pragma unroll
  for (int nt = 0; nt < 8; ++nt)
    #pragma unroll
    for (int r = 0; r < 8; ++r) acc[nt][r] += bv[nt];

  if (LN) {
    // C layout: slot r, half lh holds row m = lh*8 + r, cols nt*16 + l16.
    #pragma unroll
    for (int r = 0; r < 8; ++r) {
      float s = 0.0f, ss = 0.0f;
      #pragma unroll
      for (int nt = 0; nt < 8; ++nt) { float v = acc[nt][r]; s += v; ss += v * v; }
      #pragma unroll
      for (int m = 1; m < 16; m <<= 1) {      // reduce within the 16-lane half
        s  += __shfl_xor(s,  m, 32);
        ss += __shfl_xor(ss, m, 32);
      }
      float mean = s * (1.0f / 128.0f);
      float var  = ss * (1.0f / 128.0f) - mean * mean;
      float rs   = rsqrtf(var + 1e-5f);
      #pragma unroll
      for (int nt = 0; nt < 8; ++nt)
        acc[nt][r] = (acc[nt][r] - mean) * rs * gv[nt] + bev[nt];
    }
  }
  if (RELU) {
    #pragma unroll
    for (int nt = 0; nt < 8; ++nt)
      #pragma unroll
      for (int r = 0; r < 8; ++r) acc[nt][r] = fmaxf(acc[nt][r], 0.0f);
  }

  // ---- store (optionally fused residual / dual output) ----
  #pragma unroll
  for (int r = 0; r < 8; ++r) {
    int row = tileRow + lh * 8 + r;
    if (row < M) {
      size_t base = (size_t)row * 128 + l16;
      #pragma unroll
      for (int nt = 0; nt < 8; ++nt) {
        float  v = acc[nt][r];
        size_t o = base + (size_t)nt * 16;
        if (Y2)         { Y[o] = v; Y2[o] = resid[o] + v; }
        else if (resid) { Y[o] = resid[o] + v; }
        else            { Y[o] = v; }
      }
    }
  }
}

// ---------------------------------------------------------------------------
// segment_sum: agg[recv[e]] += e_new[e]  (float4 loads, f32 global atomics;
// agg is 25.6 MB -> L2-resident, atomics resolve at L2)
// ---------------------------------------------------------------------------
__global__ __launch_bounds__(256)
void segsum_kernel(const float* __restrict__ enew, const int* __restrict__ recv,
                   float* __restrict__ agg, int E)
{
  long long t = (long long)blockIdx.x * 256 + threadIdx.x;
  if (t >= (long long)E * 32) return;
  int edge = (int)(t >> 5);
  int c    = (int)(t & 31) * 4;
  f32x4 v = *(const f32x4*)&enew[(size_t)edge * 128 + c];
  float* dst = &agg[(size_t)recv[edge] * 128 + c];
  __hip_atomic_fetch_add(dst + 0, v.x, __ATOMIC_RELAXED, __HIP_MEMORY_SCOPE_AGENT);
  __hip_atomic_fetch_add(dst + 1, v.y, __ATOMIC_RELAXED, __HIP_MEMORY_SCOPE_AGENT);
  __hip_atomic_fetch_add(dst + 2, v.z, __ATOMIC_RELAXED, __HIP_MEMORY_SCOPE_AGENT);
  __hip_atomic_fetch_add(dst + 3, v.w, __ATOMIC_RELAXED, __HIP_MEMORY_SCOPE_AGENT);
}

// ---------------------------------------------------------------------------
// Decoder output layer: out[M,3] = H[M,128] @ Wout[128,3] + bout  (N=3: VALU)
// ---------------------------------------------------------------------------
__global__ __launch_bounds__(256)
void dec_out_kernel(const float* __restrict__ H, const float* __restrict__ Wout,
                    const float* __restrict__ bout, float* __restrict__ out, int M)
{
  int row = blockIdx.x * 256 + threadIdx.x;
  if (row >= M) return;
  const float* h = H + (size_t)row * 128;
  float s0 = bout[0], s1 = bout[1], s2 = bout[2];
  #pragma unroll 4
  for (int k = 0; k < 128; ++k) {
    float x = h[k];
    s0 += x * Wout[k * 3 + 0];
    s1 += x * Wout[k * 3 + 1];
    s2 += x * Wout[k * 3 + 2];
  }
  out[row * 3 + 0] = s0;
  out[row * 3 + 1] = s1;
  out[row * 3 + 2] = s2;
}

// ---------------------------------------------------------------------------
// Host side
// ---------------------------------------------------------------------------
struct MLP { const float *W0,*b0,*g0,*be0,*W1,*b1,*g1,*be1,*Wo,*bo; };

static MLP take_mlp(void* const* d, int& i, bool ln) {
  MLP m{};
  m.W0 = (const float*)d[i++]; m.b0 = (const float*)d[i++];
  if (ln) { m.g0 = (const float*)d[i++]; m.be0 = (const float*)d[i++]; }
  m.W1 = (const float*)d[i++]; m.b1 = (const float*)d[i++];
  if (ln) { m.g1 = (const float*)d[i++]; m.be1 = (const float*)d[i++]; }
  m.Wo = (const float*)d[i++]; m.bo = (const float*)d[i++];
  return m;
}

extern "C" void kernel_launch(void* const* d_in, const int* in_sizes, int n_in,
                              void* d_out, int out_size, void* d_ws, size_t ws_size,
                              hipStream_t stream)
{
  const float* nodes     = (const float*)d_in[0];
  const float* edges     = (const float*)d_in[1];
  const int*   senders   = (const int*)d_in[2];
  const int*   receivers = (const int*)d_in[3];
  const int Nn = in_sizes[0] / 128;
  const int E  = in_sizes[2];

  // Param leaves in setup_inputs() insertion order:
  // enc_node(10), enc_edge(10), 10x{edge(10),node(10)}, dec(6)
  int i = 4;
  MLP enc_node = take_mlp(d_in, i, true);
  MLP enc_edge = take_mlp(d_in, i, true);
  MLP pe[10], pn[10];
  for (int s = 0; s < 10; ++s) { pe[s] = take_mlp(d_in, i, true); pn[s] = take_mlp(d_in, i, true); }
  MLP dec = take_mlp(d_in, i, false);

  float* ws    = (float*)d_ws;
  float* n_lat = ws;                          // [Nn,128]
  float* aggb  = n_lat + (size_t)Nn * 128;    // [Nn,128]
  float* e_lat = aggb  + (size_t)Nn * 128;    // [E,128]
  float* e_new = e_lat + (size_t)E  * 128;    // [E,128]
  float* h     = e_new + (size_t)E  * 128;    // [E,128] hidden scratch
  __bf16* wbase = (__bf16*)(h + (size_t)E * 128);

  const dim3 blk(256);
  auto grid = [](int M) { return dim3((unsigned)((M + 127) / 128)); };

  // ---- prep: convert+swizzle every GEMM weight matrix to bf16 images ----
  size_t woff = 0;
  auto swz = [&](const float* W, int kdim) -> const __bf16* {
    __bf16* dst = wbase + woff;
    woff += (size_t)kdim * 128;
    int total = kdim * 16;                    // threads (8 elems each)
    swizzle_w_kernel<<<dim3((unsigned)((total + 255) / 256)), blk, 0, stream>>>(W, dst, kdim);
    return dst;
  };
  const __bf16 *enW0 = swz(enc_node.W0,128), *enW1 = swz(enc_node.W1,128), *enWo = swz(enc_node.Wo,128);
  const __bf16 *eeW0 = swz(enc_edge.W0,128), *eeW1 = swz(enc_edge.W1,128), *eeWo = swz(enc_edge.Wo,128);
  const __bf16 *peW0[10], *peW1[10], *peWo[10], *pnW0[10], *pnW1[10], *pnWo[10];
  for (int s = 0; s < 10; ++s) {
    peW0[s] = swz(pe[s].W0, 384); peW1[s] = swz(pe[s].W1, 128); peWo[s] = swz(pe[s].Wo, 128);
    pnW0[s] = swz(pn[s].W0, 256); pnW1[s] = swz(pn[s].W1, 128); pnWo[s] = swz(pn[s].Wo, 128);
  }
  const __bf16 *dcW0 = swz(dec.W0, 128), *dcW1 = swz(dec.W1, 128);

  // ---- encode nodes ----
  mlp_layer_kernel<128,true,true><<<grid(Nn),blk,0,stream>>>(nodes,nullptr,nullptr,nullptr,nullptr,
      enW0,enc_node.b0,enc_node.g0,enc_node.be0, h,nullptr,nullptr, Nn);
  mlp_layer_kernel<128,true,true><<<grid(Nn),blk,0,stream>>>(h,nullptr,nullptr,nullptr,nullptr,
      enW1,enc_node.b1,enc_node.g1,enc_node.be1, h,nullptr,nullptr, Nn);
  mlp_layer_kernel<128,false,false><<<grid(Nn),blk,0,stream>>>(h,nullptr,nullptr,nullptr,nullptr,
      enWo,enc_node.bo,nullptr,nullptr, n_lat,nullptr,nullptr, Nn);

  // ---- encode edges ----
  mlp_layer_kernel<128,true,true><<<grid(E),blk,0,stream>>>(edges,nullptr,nullptr,nullptr,nullptr,
      eeW0,enc_edge.b0,enc_edge.g0,enc_edge.be0, h,nullptr,nullptr, E);
  mlp_layer_kernel<128,true,true><<<grid(E),blk,0,stream>>>(h,nullptr,nullptr,nullptr,nullptr,
      eeW1,enc_edge.b1,enc_edge.g1,enc_edge.be1, h,nullptr,nullptr, E);
  mlp_layer_kernel<128,false,false><<<grid(E),blk,0,stream>>>(h,nullptr,nullptr,nullptr,nullptr,
      eeWo,enc_edge.bo,nullptr,nullptr, e_lat,nullptr,nullptr, E);

  // ---- 10 InteractionNetwork steps ----
  for (int s = 0; s < 10; ++s) {
    // edge MLP on concat(n[snd], n[rcv], e) gathered on the fly
    mlp_layer_kernel<384,true,true><<<grid(E),blk,0,stream>>>(n_lat,senders, n_lat,receivers, e_lat,
        peW0[s],pe[s].b0,pe[s].g0,pe[s].be0, h,nullptr,nullptr, E);
    mlp_layer_kernel<128,true,true><<<grid(E),blk,0,stream>>>(h,nullptr,nullptr,nullptr,nullptr,
        peW1[s],pe[s].b1,pe[s].g1,pe[s].be1, h,nullptr,nullptr, E);
    // out layer: e_new = y,  e = e + y  (fused dual store)
    mlp_layer_kernel<128,false,false><<<grid(E),blk,0,stream>>>(h,nullptr,nullptr,nullptr,nullptr,
        peWo[s],pe[s].bo,nullptr,nullptr, e_new, e_lat, e_lat, E);

    hipMemsetAsync(aggb, 0, (size_t)Nn * 128 * sizeof(float), stream);
    {
      long long tt = (long long)E * 32;
      segsum_kernel<<<dim3((unsigned)((tt + 255) / 256)), blk, 0, stream>>>(e_new, receivers, aggb, E);
    }

    // node MLP on concat(n, agg) gathered on the fly; out layer fuses n += y
    mlp_layer_kernel<256,true,true><<<grid(Nn),blk,0,stream>>>(n_lat,nullptr, aggb,nullptr, nullptr,
        pnW0[s],pn[s].b0,pn[s].g0,pn[s].be0, h,nullptr,nullptr, Nn);
    mlp_layer_kernel<128,true,true><<<grid(Nn),blk,0,stream>>>(h,nullptr,nullptr,nullptr,nullptr,
        pnW1[s],pn[s].b1,pn[s].g1,pn[s].be1, h,nullptr,nullptr, Nn);
    mlp_layer_kernel<128,false,false><<<grid(Nn),blk,0,stream>>>(h,nullptr,nullptr,nullptr,nullptr,
        pnWo[s],pn[s].bo,nullptr,nullptr, n_lat, n_lat, nullptr, Nn);
  }

  // ---- decode ----
  mlp_layer_kernel<128,false,true><<<grid(Nn),blk,0,stream>>>(n_lat,nullptr,nullptr,nullptr,nullptr,
      dcW0,dec.b0,nullptr,nullptr, h,nullptr,nullptr, Nn);
  mlp_layer_kernel<128,false,true><<<grid(Nn),blk,0,stream>>>(h,nullptr,nullptr,nullptr,nullptr,
      dcW1,dec.b1,nullptr,nullptr, h,nullptr,nullptr, Nn);
  dec_out_kernel<<<dim3((unsigned)((Nn + 255) / 256)), blk, 0, stream>>>(h, dec.Wo, dec.bo, (float*)d_out, Nn);
}
